// lstm_seq2seq_43980465111341
// MI455X (gfx1250) — matrix operand order repaired and checked by reading, not compiled, not run on hardware
//
#include <hip/hip_runtime.h>
#include <hip/hip_bf16.h>

typedef __bf16 bf16_t;
typedef bf16_t v16bf __attribute__((ext_vector_type(16)));
typedef float  v8f   __attribute__((ext_vector_type(8)));

#define SEQ_LEN 64
#define BATCH   1024
#define HID     512
#define GATES   2048
#define TGT     32
#define APAD    40      // padded A-tile row stride in bf16 elements (80 B)

// ---------------------------------------------------------------------------
// Pack [Whh (G x 512) | Wih (G x Kxa, zero-padded)] into bf16 WMMA-B tiles.
// Tile (jt, kt): 32 lanes x 16 bf16, linear = ((jt*nkt + kt)*32 + lane)*16 + e
// where j = jt*16 + (lane&15), k(in tile) = (lane>>4)*16 + e.
// ---------------------------------------------------------------------------
__global__ __launch_bounds__(256) void pack_w_kernel(
    const float* __restrict__ Whh, const float* __restrict__ Wih,
    int Kxa, int nkt, bf16_t* __restrict__ dst)
{
    int tid = blockIdx.x * blockDim.x + threadIdx.x;
    int total = GATES * nkt * 32;          // elements (G * Kc)
    if (tid >= total) return;
    int e       = tid & 15;
    int lane    = (tid >> 4) & 31;
    int tileIdx = tid >> 9;
    int kt = tileIdx % nkt;
    int jt = tileIdx / nkt;
    int j  = jt * 16 + (lane & 15);
    int k  = kt * 32 + ((lane >> 4) << 4) + e;
    float v = 0.0f;
    if (k < HID) {
        v = Whh[j * HID + k];
    } else {
        int kk = k - HID;
        if (kk < Kxa) v = Wih[j * Kxa + kk];
    }
    dst[tid] = (bf16_t)v;
}

__global__ __launch_bounds__(256) void pack_bias_kernel(
    const float* __restrict__ a, const float* __restrict__ b, float* __restrict__ dst)
{
    int i = blockIdx.x * blockDim.x + threadIdx.x;
    if (i < GATES) dst[i] = a[i] + b[i];
}

__global__ __launch_bounds__(256) void zero_kernel(float* __restrict__ p, int n)
{
    int i = blockIdx.x * blockDim.x + threadIdx.x;
    int stride = gridDim.x * blockDim.x;
    for (; i < n; i += stride) p[i] = 0.0f;
}

__global__ __launch_bounds__(256) void dec_seed_kernel(
    const float* __restrict__ x, float* __restrict__ fb)
{
    int i = blockIdx.x * blockDim.x + threadIdx.x;
    if (i < BATCH * 2) {
        int b = i >> 1, j = i & 1;
        fb[i] = x[(63 * BATCH + b) * 4 + j];   // x[-1, :, :2]
    }
}

// ---------------------------------------------------------------------------
// Fused LSTM cell: gates = [h|x] @ [Whh|Wih]^T + (bih+bhh); then elementwise.
// Block: 256 threads (8 waves). Tile: M=64 batch rows, N=64 hidden cols, all
// 4 gate groups. Wave w: mt = w&3 (16 rows), ntp = w>>2 (two 16-col n-tiles),
// 8 accumulators (4 groups x 2 n-tiles). K loop in chunks of 32 (bf16 WMMA),
// double-buffered LDS (A staged as bf16 with padded stride, B pre-swizzled).
// FULLX=true: x input is a full 512-wide activation (branchless loads).
// ---------------------------------------------------------------------------
template <bool FULLX>
__global__ __launch_bounds__(256) void lstm_cell_kernel(
    const float* __restrict__ h_in,
    const float* __restrict__ x_in, int ldx, int Kxa,
    const bf16_t* __restrict__ Wp, int nkt,
    const float* __restrict__ bias,
    float* __restrict__ c_io,
    float* __restrict__ h_out)
{
    __shared__ __align__(16) bf16_t ldsA[2][64 * APAD];   // 2 x 5 KB bf16
    __shared__ __align__(64) bf16_t ldsB[2][16 * 512];    // 2 x 16 KB tiles

    const int tid  = threadIdx.x;
    const int lane = tid & 31;
    const int w    = tid >> 5;
    const int ntp  = w >> 2;                 // 0..1
    const int m_blk = blockIdx.x * 64;
    const int n_off = blockIdx.y * 64;
    const int jt_base = n_off >> 4;

    // ---- staging helpers ----
    auto stageA = [&](int kt, int buf) {
        const bool  isH   = (kt < 16);
        const float* src  = isH ? h_in : x_in;
        const int    ld   = isH ? HID : ldx;
        const int    kb   = isH ? kt * 32 : (kt - 16) * 32;
#pragma unroll
        for (int r = 0; r < 2; ++r) {
            int f4  = tid + 256 * r;            // 0..511
            int row = f4 >> 3;
            int col = (f4 & 7) * 4;
            float4 v;
            if (FULLX || isH) {
                v = *(const float4*)(src + (m_blk + row) * ld + kb + col);
            } else {
                const float* rp = src + (m_blk + row) * ld;
                v.x = (kb + col + 0 < Kxa) ? rp[kb + col + 0] : 0.0f;
                v.y = (kb + col + 1 < Kxa) ? rp[kb + col + 1] : 0.0f;
                v.z = (kb + col + 2 < Kxa) ? rp[kb + col + 2] : 0.0f;
                v.w = (kb + col + 3 < Kxa) ? rp[kb + col + 3] : 0.0f;
            }
            bf16_t* d = &ldsA[buf][row * APAD + col];
            d[0] = (bf16_t)v.x; d[1] = (bf16_t)v.y;
            d[2] = (bf16_t)v.z; d[3] = (bf16_t)v.w;
        }
    };
    auto stageB = [&](int kt, int buf) {
        int tile = tid >> 4;                 // 0..15: g*4 + nt
        int g  = tile >> 2;
        int nt = tile & 3;
        int jt = g * 32 + jt_base + nt;
        const bf16_t* tsrc = Wp + (size_t)(jt * nkt + kt) * 512;
        bf16_t* tdst = &ldsB[buf][tile * 512];
        int t4 = tid & 15;
#pragma unroll
        for (int u = 0; u < 4; ++u) {
            int eo = t4 * 32 + u * 8;        // 8 bf16 = 16 B
            *(uint4*)(tdst + eo) = *(const uint4*)(tsrc + eo);
        }
    };

    // Accumulators pre-seeded with bias (bias added before GEMM == after).
    v8f acc[8];
#pragma unroll
    for (int g = 0; g < 4; ++g)
#pragma unroll
        for (int ntl = 0; ntl < 2; ++ntl) {
            int n = n_off + (ntp * 2 + ntl) * 16 + (lane & 15);
            float bv = bias[g * HID + n];
            v8f t;
#pragma unroll
            for (int q = 0; q < 8; ++q) t[q] = bv;
            acc[g * 2 + ntl] = t;
        }

    stageA(0, 0);
    stageB(0, 0);

    for (int kt = 0; kt < nkt; ++kt) {
        __syncthreads();
        const int cur = kt & 1;
        if (kt + 1 < nkt) {                  // overlap next stage with WMMAs
            stageA(kt + 1, cur ^ 1);
            stageB(kt + 1, cur ^ 1);
        }

        // ---- A fragment: contiguous k order (lane half selects K half, as in B) --
        v16bf afrag;
        {
            int row = (w & 3) * 16 + (lane & 15);
            int kb  = (lane >> 4) * 16;
            const bf16_t* ap = &ldsA[cur][row * APAD];
            union { uint4 u[2]; v16bf v; } au;
            au.u[0] = *(const uint4*)(ap + kb);        // k = kb .. kb+7
            au.u[1] = *(const uint4*)(ap + kb + 8);    // k = kb+8 .. kb+15
            afrag = au.v;
        }

        // ---- 8 WMMAs: 4 gate groups x 2 n-tiles ----
#pragma unroll
        for (int g = 0; g < 4; ++g)
#pragma unroll
            for (int ntl = 0; ntl < 2; ++ntl) {
                int tile = g * 4 + (ntp * 2 + ntl);
                v16bf bfrag = *(const v16bf*)(&ldsB[cur][tile * 512 + lane * 16]);
                acc[g * 2 + ntl] = __builtin_amdgcn_wmma_f32_16x16x32_bf16(
                    false, afrag, false, bfrag, (short)0, acc[g * 2 + ntl],
                    false, false);
            }
    }

    // ---- epilogue: i,f,g,o are element-aligned across the 4 group tiles ----
    int mrow = m_blk + (w & 3) * 16 + ((lane >> 4) << 3);
#pragma unroll
    for (int ntl = 0; ntl < 2; ++ntl) {
        int n = n_off + (ntp * 2 + ntl) * 16 + (lane & 15);
#pragma unroll
        for (int q = 0; q < 8; ++q) {
            int m = mrow + q;
            float iv = acc[0 * 2 + ntl][q];
            float fv = acc[1 * 2 + ntl][q];
            float gv = acc[2 * 2 + ntl][q];
            float ov = acc[3 * 2 + ntl][q];
            float si = 1.0f / (1.0f + __expf(-iv));
            float sf = 1.0f / (1.0f + __expf(-fv));
            float so = 1.0f / (1.0f + __expf(-ov));
            float tg = tanhf(gv);
            int idx = m * HID + n;
            float cn = sf * c_io[idx] + si * tg;
            c_io[idx]  = cn;
            h_out[idx] = so * tanhf(cn);
        }
    }
}

// out[b, 0:2] = h[b,:] @ lin_W^T + lin_b ; also written to feedback buffer.
__global__ __launch_bounds__(256) void linear_kernel(
    const float* __restrict__ h, const float* __restrict__ W,
    const float* __restrict__ b, float* __restrict__ out, float* __restrict__ fb)
{
    int bi = blockIdx.x * blockDim.x + threadIdx.x;
    if (bi >= BATCH) return;
    float a0 = b[0], a1 = b[1];
    const float* hr = h + bi * HID;
    for (int k = 0; k < HID; k += 4) {
        float4 hv = *(const float4*)(hr + k);
        float4 w0 = *(const float4*)(W + k);
        float4 w1 = *(const float4*)(W + HID + k);
        a0 += hv.x * w0.x + hv.y * w0.y + hv.z * w0.z + hv.w * w0.w;
        a1 += hv.x * w1.x + hv.y * w1.y + hv.z * w1.z + hv.w * w1.w;
    }
    out[bi * 2 + 0] = a0; out[bi * 2 + 1] = a1;
    fb[bi * 2 + 0]  = a0; fb[bi * 2 + 1]  = a1;
}

// ---------------------------------------------------------------------------
extern "C" void kernel_launch(void* const* d_in, const int* in_sizes, int n_in,
                              void* d_out, int out_size, void* d_ws, size_t ws_size,
                              hipStream_t stream)
{
    const float* x        = (const float*)d_in[0];
    const float* eWih0    = (const float*)d_in[1];
    const float* eWhh0    = (const float*)d_in[2];
    const float* ebih0    = (const float*)d_in[3];
    const float* ebhh0    = (const float*)d_in[4];
    const float* eWih     = (const float*)d_in[5];
    const float* eWhh     = (const float*)d_in[6];
    const float* ebih     = (const float*)d_in[7];
    const float* ebhh     = (const float*)d_in[8];
    const float* dWih0    = (const float*)d_in[9];
    const float* dWhh0    = (const float*)d_in[10];
    const float* dbih0    = (const float*)d_in[11];
    const float* dbhh0    = (const float*)d_in[12];
    const float* dWih     = (const float*)d_in[13];
    const float* dWhh     = (const float*)d_in[14];
    const float* dbih     = (const float*)d_in[15];
    const float* dbhh     = (const float*)d_in[16];
    const float* linW     = (const float*)d_in[17];
    const float* linB     = (const float*)d_in[18];
    float* out = (float*)d_out;

    // ---- workspace carve-up (256 B aligned) ----
    size_t off = 0;
    auto take = [&](size_t bytes) -> void* {
        void* p = (char*)d_ws + off;
        off += (bytes + 255) & ~(size_t)255;
        return p;
    };
    const int nktE[3] = {17, 32, 32};
    const int nktD[3] = {17, 32, 32};
    bf16_t* WE[3]; bf16_t* WD[3];
    for (int l = 0; l < 3; ++l) WE[l] = (bf16_t*)take((size_t)GATES * nktE[l] * 32 * 2);
    for (int l = 0; l < 3; ++l) WD[l] = (bf16_t*)take((size_t)GATES * nktD[l] * 32 * 2);
    float* BE[3]; float* BD[3];
    for (int l = 0; l < 3; ++l) BE[l] = (float*)take(GATES * 4);
    for (int l = 0; l < 3; ++l) BD[l] = (float*)take(GATES * 4);
    const size_t SB = (size_t)BATCH * HID;           // one state buffer, floats
    float* states = (float*)take(9 * SB * 4);        // h[3][2] then c[3]
    float* fb     = (float*)take(BATCH * 2 * 4);     // decoder input feedback
    if (off > ws_size) return;                       // insufficient scratch

    auto hbuf = [&](int l, int p) { return states + (size_t)(l * 2 + p) * SB; };
    auto cbuf = [&](int l)        { return states + (size_t)(6 + l) * SB; };

    // ---- prep: pack weights, biases, zero states, seed decoder input ----
    {
        const float* WihE[3] = {eWih0, eWih, eWih + (size_t)GATES * HID};
        const float* WhhE[3] = {eWhh0, eWhh, eWhh + (size_t)GATES * HID};
        const float* WihD[3] = {dWih0, dWih, dWih + (size_t)GATES * HID};
        const float* WhhD[3] = {dWhh0, dWhh, dWhh + (size_t)GATES * HID};
        const int KxaE[3] = {4, 512, 512};
        const int KxaD[3] = {2, 512, 512};
        for (int l = 0; l < 3; ++l) {
            int total = GATES * nktE[l] * 32;
            pack_w_kernel<<<(total + 255) / 256, 256, 0, stream>>>(
                WhhE[l], WihE[l], KxaE[l], nktE[l], WE[l]);
            total = GATES * nktD[l] * 32;
            pack_w_kernel<<<(total + 255) / 256, 256, 0, stream>>>(
                WhhD[l], WihD[l], KxaD[l], nktD[l], WD[l]);
        }
        const float* bihE[3] = {ebih0, ebih, ebih + GATES};
        const float* bhhE[3] = {ebhh0, ebhh, ebhh + GATES};
        const float* bihD[3] = {dbih0, dbih, dbih + GATES};
        const float* bhhD[3] = {dbhh0, dbhh, dbhh + GATES};
        for (int l = 0; l < 3; ++l) {
            pack_bias_kernel<<<(GATES + 255) / 256, 256, 0, stream>>>(bihE[l], bhhE[l], BE[l]);
            pack_bias_kernel<<<(GATES + 255) / 256, 256, 0, stream>>>(bihD[l], bhhD[l], BD[l]);
        }
        zero_kernel<<<1024, 256, 0, stream>>>(states, (int)(9 * SB));
        dec_seed_kernel<<<(BATCH * 2 + 255) / 256, 256, 0, stream>>>(x, fb);
    }

    const dim3 cgrid(BATCH / 64, HID / 64);   // 16 x 8

    // ---- encoder: 64 steps x 3 layers ----
    for (int t = 0; t < SEQ_LEN; ++t) {
        int p = t & 1;
        // layer 0: tiny x (width 4), guarded loads
        lstm_cell_kernel<false><<<cgrid, 256, 0, stream>>>(
            hbuf(0, p), x + (size_t)t * BATCH * 4, 4, 4,
            WE[0], nktE[0], BE[0], cbuf(0), hbuf(0, 1 - p));
        // layers 1..2: full-width x, branchless
        for (int l = 1; l < 3; ++l) {
            lstm_cell_kernel<true><<<cgrid, 256, 0, stream>>>(
                hbuf(l, p), hbuf(l - 1, 1 - p), HID, HID,
                WE[l], nktE[l], BE[l], cbuf(l), hbuf(l, 1 - p));
        }
    }
    // encoder t=63 wrote parity 0 -> decoder s=0 reads parity 0.

    // ---- decoder: 32 steps x (3 layers + linear head) ----
    for (int s = 0; s < TGT; ++s) {
        int p = s & 1;
        lstm_cell_kernel<false><<<cgrid, 256, 0, stream>>>(
            hbuf(0, p), fb, 2, 2,
            WD[0], nktD[0], BD[0], cbuf(0), hbuf(0, 1 - p));
        for (int l = 1; l < 3; ++l) {
            lstm_cell_kernel<true><<<cgrid, 256, 0, stream>>>(
                hbuf(l, p), hbuf(l - 1, 1 - p), HID, HID,
                WD[l], nktD[l], BD[l], cbuf(l), hbuf(l, 1 - p));
        }
        linear_kernel<<<(BATCH + 255) / 256, 256, 0, stream>>>(
            hbuf(2, 1 - p), linW, linB, out + (size_t)s * BATCH * 2, fb);
    }
}
